// GCN_33036888441507
// MI455X (gfx1250) — matrix operand order repaired and checked by reading, not compiled, not run on hardware
//
#include <hip/hip_runtime.h>
#include <stdint.h>

typedef __attribute__((ext_vector_type(4)))  float          f32x4;
typedef __attribute__((ext_vector_type(8)))  float          f32x8;
typedef __attribute__((ext_vector_type(8)))  unsigned short u16x8;
typedef __attribute__((ext_vector_type(16))) unsigned short u16x16;
typedef __attribute__((ext_vector_type(16))) __bf16         bf16x16;

#define NL     4096
#define NB     4
#define NROWS  (NB * NL)       // 16384
#define LRELU  0.01f

__device__ __forceinline__ unsigned short f2bf(float f) {
  unsigned int u = __float_as_uint(f);
  u += 0x7FFFu + ((u >> 16) & 1u);          // round-to-nearest-even
  return (unsigned short)(u >> 16);
}
__device__ __forceinline__ float bf2f(unsigned short h) {
  return __uint_as_float(((unsigned int)h) << 16);
}

// ---- rowsum(A) -> dinv = 1/sqrt(rowsum), once (A_hat identical across layers)
__global__ void k_dinv(const float* __restrict__ A, float* __restrict__ dinv) {
  int row = blockIdx.x;                               // 0..16383 (b*4096+i)
  const float* ar = A + (size_t)row * NL;
  float s = 0.f;
  for (int k = threadIdx.x; k < NL; k += 256) s += ar[k];
  for (int off = 16; off; off >>= 1) s += __shfl_down(s, off, 32);
  __shared__ float red[8];
  int lane = threadIdx.x & 31, w = threadIdx.x >> 5;
  if (lane == 0) red[w] = s;
  __syncthreads();
  if (threadIdx.x == 0) {
    float t = 0.f;
    for (int i = 0; i < 8; ++i) t += red[i];
    dinv[row] = (float)(1.0 / sqrt((double)t));
  }
}

// ---- BN stats stage 1: per-block partial sums (deterministic, no atomics)
__global__ void k_bnstats(const float* __restrict__ H, int c, double* __restrict__ part) {
  int ch = threadIdx.x;                  // 256 threads, ch < c active
  int row0 = blockIdx.x * 64;            // 256 blocks x 64 rows
  double s = 0.0, q = 0.0;
  if (ch < c) {
    for (int r = 0; r < 64; ++r) {
      float x = H[(size_t)(row0 + r) * c + ch];
      s += x; q += (double)x * x;
    }
  }
  part[(size_t)blockIdx.x * 512 + ch]       = s;
  part[(size_t)blockIdx.x * 512 + 256 + ch] = q;
}

// ---- BN stats stage 2: reduce partials, compute scale/shift
__global__ void k_bnfinalize(const double* __restrict__ part, const float* __restrict__ g,
                             const float* __restrict__ bb, int c,
                             float* __restrict__ scale, float* __restrict__ shift) {
  int ch = threadIdx.x;
  if (ch >= c) return;
  double s = 0.0, q = 0.0;
  for (int blk = 0; blk < 256; ++blk) {
    s += part[(size_t)blk * 512 + ch];
    q += part[(size_t)blk * 512 + 256 + ch];
  }
  double mu  = s / (double)NROWS;
  double var = q / (double)NROWS - mu * mu;
  float sc = g[ch] * (float)(1.0 / sqrt(var + (double)1e-5f));
  scale[ch] = sc;
  shift[ch] = bb[ch] - (float)mu * sc;
}

// ---- fold BN into W:  W'[k][n] = scale[k]*W[k][n]
__global__ void k_foldW(const float* __restrict__ W, const float* __restrict__ scale,
                        int c, int dout, float* __restrict__ Wf) {
  int t = blockIdx.x * blockDim.x + threadIdx.x;
  if (t >= c * dout) return;
  Wf[t] = W[t] * scale[t / dout];
}

// ---- fold BN shift into bias:  b'[n] = bias[n] + sum_k shift[k]*W[k][n]
__global__ void k_foldB(const float* __restrict__ W, const float* __restrict__ shift,
                        const float* __restrict__ bias, int c, int dout, float* __restrict__ bf_) {
  int col = threadIdx.x;
  if (col >= dout) return;
  float s = bias[col];
  for (int k = 0; k < c; ++k) s = fmaf(shift[k], W[k * dout + col], s);
  bf_[col] = s;
}

// ---- dense GEMM: HW = Hn @ W' + b' ; also emit transposed bf16 hi/lo for WMMA B
__global__ void k_dense(const float* __restrict__ H, const float* __restrict__ Wf,
                        const float* __restrict__ bf_, int din, int dout,
                        float* __restrict__ HW, unsigned short* __restrict__ Thi,
                        unsigned short* __restrict__ Tlo) {
  int t   = blockIdx.x * blockDim.x + threadIdx.x;  // NROWS*dout threads
  int col = t % dout;
  int row = t / dout;                               // wave-uniform (dout >= 32)
  const float* hr = H + (size_t)row * din;
  float acc = bf_[col];
  for (int k = 0; k < din; ++k) acc = fmaf(hr[k], Wf[k * dout + col], acc);
  HW[(size_t)row * dout + col] = acc;
  unsigned short hi = f2bf(acc);
  unsigned short lo = f2bf(acc - bf2f(hi));
  int b = row >> 12, r = row & (NL - 1);
  size_t o = ((size_t)b * dout + col) * NL + r;     // [b][n][k] layout
  Thi[o] = hi; Tlo[o] = lo;
}

// ---- main WMMA kernel: Hnext = leakyrelu( (dinv_i*A*dinv_j) @ HW + HW )
template <int NT>
__global__ void __launch_bounds__(128)
k_spmm(const float* __restrict__ A, const float* __restrict__ dinv,
       const float* __restrict__ HW, const unsigned short* __restrict__ Thi,
       const unsigned short* __restrict__ Tlo, float* __restrict__ Hn) {
  const int dout = NT * 16;
  int wave = blockIdx.x * 4 + (threadIdx.x >> 5);   // 1024 waves total
  int b  = wave >> 8;                               // 256 M-tiles per batch
  int mt = wave & 255;
  int i0 = mt << 4;
  int lane = threadIdx.x & 31;
  int lrow = lane & 15, lhi = lane >> 4;

  const float* Ab  = A    + (size_t)b * NL * NL;
  const float* db  = dinv + (size_t)b * NL;
  const float* HWb = HW   + (size_t)b * NL * dout;
  const unsigned short* ThiB = Thi + (size_t)b * dout * NL;
  const unsigned short* TloB = Tlo + (size_t)b * dout * NL;
  float* HnB = Hn + (size_t)b * NL * dout;

  int row = i0 + lrow;
  const float* ar = Ab + (size_t)row * NL;
  float di = db[row];
  int ka = lhi << 3;                                // A K-offset: 0 / 8
  int kb = lhi << 3;                                // B K-offset: 0 / 8

  f32x8 acc[NT] = {};

  for (int k0 = 0; k0 < NL; k0 += 32) {
    const float* ap = ar + k0 + ka;
    __builtin_prefetch(ap + 64, 0, 0);
    f32x4 a0 = *(const f32x4*)(ap);
    f32x4 a1 = *(const f32x4*)(ap + 4);
    f32x4 a2 = *(const f32x4*)(ap + 16);
    f32x4 a3 = *(const f32x4*)(ap + 20);
    const float* dp = db + k0 + ka;
    f32x4 d0 = *(const f32x4*)(dp);
    f32x4 d1 = *(const f32x4*)(dp + 4);
    f32x4 d2 = *(const f32x4*)(dp + 16);
    f32x4 d3 = *(const f32x4*)(dp + 20);

    u16x16 ahiU, aloU;
#pragma unroll
    for (int e = 0; e < 4; ++e) {                   // K-ascending per ISA layout
      float v0 = di * d0[e] * a0[e];
      float v1 = di * d1[e] * a1[e];
      float v2 = di * d2[e] * a2[e];
      float v3 = di * d3[e] * a3[e];
      unsigned short h;
      h = f2bf(v0); ahiU[e]      = h; aloU[e]      = f2bf(v0 - bf2f(h));
      h = f2bf(v1); ahiU[4 + e]  = h; aloU[4 + e]  = f2bf(v1 - bf2f(h));
      h = f2bf(v2); ahiU[8 + e]  = h; aloU[8 + e]  = f2bf(v2 - bf2f(h));
      h = f2bf(v3); ahiU[12 + e] = h; aloU[12 + e] = f2bf(v3 - bf2f(h));
    }
    bf16x16 ahi = __builtin_bit_cast(bf16x16, ahiU);
    bf16x16 alo = __builtin_bit_cast(bf16x16, aloU);

#pragma unroll
    for (int nt = 0; nt < NT; ++nt) {
      size_t bo = ((size_t)(nt * 16 + lrow)) * NL + (size_t)(k0 + kb);
      u16x8 b0h = *(const u16x8*)(ThiB + bo);
      u16x8 b1h = *(const u16x8*)(ThiB + bo + 16);
      u16x8 b0l = *(const u16x8*)(TloB + bo);
      u16x8 b1l = *(const u16x8*)(TloB + bo + 16);
      u16x16 bhU, blU;
#pragma unroll
      for (int e = 0; e < 8; ++e) {
        bhU[e] = b0h[e]; bhU[8 + e] = b1h[e];
        blU[e] = b0l[e]; blU[8 + e] = b1l[e];
      }
      bf16x16 bh = __builtin_bit_cast(bf16x16, bhU);
      bf16x16 bl = __builtin_bit_cast(bf16x16, blU);
      acc[nt] = __builtin_amdgcn_wmma_f32_16x16x32_bf16(false, ahi, false, bh, (short)0, acc[nt], false, false);
      acc[nt] = __builtin_amdgcn_wmma_f32_16x16x32_bf16(false, ahi, false, bl, (short)0, acc[nt], false, false);
      acc[nt] = __builtin_amdgcn_wmma_f32_16x16x32_bf16(false, alo, false, bh, (short)0, acc[nt], false, false);
    }
  }

  // epilogue: + identity term (HW), leaky relu; C-layout: vgpr r -> M=r+8*lhi, N=lrow
#pragma unroll
  for (int nt = 0; nt < NT; ++nt) {
    int col = nt * 16 + lrow;
#pragma unroll
    for (int r = 0; r < 8; ++r) {
      int orow = i0 + r + 8 * lhi;
      float v = acc[nt][r] + HWb[(size_t)orow * dout + col];
      HnB[(size_t)orow * dout + col] = v > 0.f ? v : LRELU * v;
    }
  }
}

// ---- final gather: out[b][ch][j] = H4[b][idx[b][j]][ch]
__global__ void k_gather(const float* __restrict__ H4, const int* __restrict__ idx,
                         float* __restrict__ out) {
  int t = blockIdx.x * blockDim.x + threadIdx.x;     // 4*64*1024
  if (t >= NB * 64 * 1024) return;
  int j  = t & 1023;
  int ch = (t >> 10) & 63;
  int b  = t >> 16;
  int node = idx[b * 1024 + j];
  out[t] = H4[((size_t)b * NL + node) * 64 + ch];
}

extern "C" void kernel_launch(void* const* d_in, const int* in_sizes, int n_in,
                              void* d_out, int out_size, void* d_ws, size_t ws_size,
                              hipStream_t stream) {
  (void)in_sizes; (void)n_in; (void)out_size; (void)ws_size;
  const float* x   = (const float*)d_in[0];          // [4,64,64,200] == [4,4096,200]
  const float* A   = (const float*)d_in[1];          // [4,4096,4096]
  const int*   idx = (const int*)d_in[2];            // [4,1024]
  // d_in[3] = band_patch (unused)
  static const int dins[4]  = {200, 128, 128, 128};
  static const int douts[4] = {128, 128, 128, 64};

  char* p = (char*)d_ws;
  auto carve = [&](size_t bytes) { char* q = p; p += (bytes + 255) & ~(size_t)255; return q; };
  float* dinv = (float*)carve((size_t)NROWS * 4);
  float* Hb0  = (float*)carve((size_t)NROWS * 200 * 4);
  float* Hb1  = (float*)carve((size_t)NROWS * 128 * 4);
  float* HW   = (float*)carve((size_t)NROWS * 128 * 4);
  unsigned short* Thi = (unsigned short*)carve((size_t)NROWS * 128 * 2);
  unsigned short* Tlo = (unsigned short*)carve((size_t)NROWS * 128 * 2);
  float*  Wf    = (float*)carve(200 * 128 * 4);
  float*  bf_   = (float*)carve(128 * 4);
  double* part  = (double*)carve((size_t)256 * 512 * 8);
  float*  scale = (float*)carve(256 * 4);
  float*  shift = (float*)carve(256 * 4);

  k_dinv<<<NROWS, 256, 0, stream>>>(A, dinv);

  const float* Hcur = x;
  float* Hnext = Hb0;
  for (int i = 0; i < 4; ++i) {
    const float* g    = (const float*)d_in[4 + 4 * i + 0];
    const float* bb   = (const float*)d_in[4 + 4 * i + 1];
    const float* W    = (const float*)d_in[4 + 4 * i + 2];
    const float* bias = (const float*)d_in[4 + 4 * i + 3];
    int din = dins[i], dout = douts[i];

    k_bnstats<<<256, 256, 0, stream>>>(Hcur, din, part);
    k_bnfinalize<<<1, 256, 0, stream>>>(part, g, bb, din, scale, shift);
    k_foldW<<<(din * dout + 255) / 256, 256, 0, stream>>>(W, scale, din, dout, Wf);
    k_foldB<<<1, 128, 0, stream>>>(W, shift, bias, din, dout, bf_);
    k_dense<<<(NROWS * dout) / 256, 256, 0, stream>>>(Hcur, Wf, bf_, din, dout, HW, Thi, Tlo);
    if (dout == 128) k_spmm<8><<<256, 128, 0, stream>>>(A, dinv, HW, Thi, Tlo, Hnext);
    else             k_spmm<4><<<256, 128, 0, stream>>>(A, dinv, HW, Thi, Tlo, Hnext);

    Hcur  = Hnext;
    Hnext = (Hnext == Hb0) ? Hb1 : Hb0;
  }
  k_gather<<<(NB * 64 * 1024) / 256, 256, 0, stream>>>(Hcur, idx, (float*)d_out);
}
